// Mamba2MixerSimple_83416854823030
// MI455X (gfx1250) — compile-verified, matrix operand order checked by reading, not run on hardware
//
#include <hip/hip_runtime.h>
#include <hip/hip_bf16.h>

// ---------------- problem constants ----------------
#define kB   2
#define kS   2048
#define kD   2048
#define kI   4096          // INTER
#define kH   64
#define kP   64
#define kGN  128           // G*N
#define kL   128           // chunk size
#define kNC  16            // S / L
#define kCD  4352          // CONV_DIM
#define kEPS 1e-5f

typedef __attribute__((ext_vector_type(16))) __bf16 v16bf;
typedef __attribute__((ext_vector_type(8)))  __bf16 v8bf;
typedef __attribute__((ext_vector_type(8)))  float  v8f;
typedef __attribute__((ext_vector_type(4)))  float  v4f;
typedef __attribute__((ext_vector_type(4)))  unsigned short v4u16;
typedef __attribute__((ext_vector_type(4)))  int    v4i_t;

// ---------------- helpers ----------------
__device__ __forceinline__ unsigned short f2bf(float f) {
    unsigned int u = __float_as_uint(f);
    u = (u + 0x7FFFu + ((u >> 16) & 1u)) >> 16;   // RNE
    return (unsigned short)u;
}
__device__ __forceinline__ float bf2f(unsigned short h) {
    return __uint_as_float(((unsigned int)h) << 16);
}

// --------- CDNA5 async global->LDS copy (16B per lane), with sync fallback ---------
#if __has_builtin(__builtin_amdgcn_global_load_async_to_lds_b128) && \
    __has_builtin(__builtin_amdgcn_s_wait_asynccnt)
#define HAS_ASYNC 1
__device__ __forceinline__ void async_cp16(void* lds, const void* g) {
    __attribute__((address_space(1))) v4i_t* gp =
        (__attribute__((address_space(1))) v4i_t*)(unsigned long long)(g);
    __attribute__((address_space(3))) v4i_t* lp =
        (__attribute__((address_space(3))) v4i_t*)(unsigned int)(unsigned long long)(lds);
    __builtin_amdgcn_global_load_async_to_lds_b128(gp, lp, 0, 0);
}
template<int N> __device__ __forceinline__ void wait_async() {
    __builtin_amdgcn_s_wait_asynccnt((unsigned short)N);
}
#else
#define HAS_ASYNC 0
__device__ __forceinline__ void async_cp16(void* lds, const void* g) {
    *(uint4*)lds = *(const uint4*)g;        // global_load_b128 + ds_store_b128
}
template<int N> __device__ __forceinline__ void wait_async() {}
#endif

// Load a 16x32 bf16 WMMA operand fragment from an LDS tile stored as [row][k].
//   lanes 0-15 : row = lane,    K = kk+{0..7}  and kk+{16..23}
//   lanes 16-31: row = lane-16, K = kk+{8..15} and kk+{24..31}
__device__ __forceinline__ v16bf load_frag(const unsigned short* base, int stride,
                                           int lane, int kk) {
    int half = (lane >> 4) & 1;
    int row  = lane & 15;
    const unsigned short* p = base + row * stride + kk + half * 8;
    v8bf lo = *(const v8bf*)(p);
    v8bf hi = *(const v8bf*)(p + 16);
    return __builtin_shufflevector(lo, hi, 0,1,2,3,4,5,6,7,8,9,10,11,12,13,14,15);
}

__device__ __forceinline__ v8f wmma_bf16(v16bf a, v16bf b, v8f c) {
    return __builtin_amdgcn_wmma_f32_16x16x32_bf16(false, a, false, b,
                                                   (short)0, c, false, false);
}

// ---------------- pre-pass: fp32 -> bf16 cast ----------------
__global__ __launch_bounds__(256) void cast_bf16_kernel(
    const float* __restrict__ src, unsigned short* __restrict__ dst, size_t n4)
{
    size_t id = (size_t)blockIdx.x * 256 + threadIdx.x;
    if (id >= n4) return;
    v4f v = *(const v4f*)(src + id * 4);
    v4u16 o = { f2bf(v[0]), f2bf(v[1]), f2bf(v[2]), f2bf(v[3]) };
    *(v4u16*)(dst + id * 4) = o;
}

// ---------------- pre-pass: W (KxN fp32) -> WT (NxK bf16) ----------------
__global__ __launch_bounds__(256) void transpose_bf16_kernel(
    const float* __restrict__ W, unsigned short* __restrict__ WT, int K, int N)
{
    __shared__ float tile[32][33];
    int k0 = blockIdx.y * 32, n0 = blockIdx.x * 32;
    int tr = threadIdx.x >> 5, tc = threadIdx.x & 31;
    #pragma unroll
    for (int j = 0; j < 4; ++j)
        tile[tr + j * 8][tc] = W[(size_t)(k0 + tr + j * 8) * N + n0 + tc];
    __syncthreads();
    #pragma unroll
    for (int j = 0; j < 4; ++j)
        WT[(size_t)(n0 + tr + j * 8) * K + k0 + tc] = f2bf(tile[tc][tr + j * 8]);
}

// ---------------- bf16 WMMA GEMM: 256x128 block, 64x64 wave tile ----------------
// C[m, coff+n] = sum_k A[m,k] * BT[n,k]
// A: MxK bf16 (lda), BT: NxK bf16 (ldbt), C fp32 (ldc).
// M % 256 == 0, K % 64 == 0. K-step 64, triple-buffered async staging,
// one barrier per K-step. 256 threads = 8 waves (4 over M x 2 over N).
__global__ __launch_bounds__(256) void gemm_bf16_kernel(
    const unsigned short* __restrict__ A, const unsigned short* __restrict__ BT,
    float* __restrict__ C,
    int M, int N, int K, int lda, int ldbt, int ldc, int coff)
{
    __shared__ __align__(16) unsigned short As[3][256 * 64];   // 96 KB
    __shared__ __align__(16) unsigned short Bs[3][128 * 64];   // 48 KB

    const int tid  = threadIdx.x;
    const int lane = tid & 31;
    const int wave = tid >> 5;
    const int wm   = wave & 3;      // 4 waves over 256 rows -> 64 rows each
    const int wn   = wave >> 2;     // 2 waves over 128 cols -> 64 cols each
    const int tileM = blockIdx.y * 256;
    const int tileN = blockIdx.x * 128;

    v8f acc[4][4];
    const v8f vzero = {0.f,0.f,0.f,0.f,0.f,0.f,0.f,0.f};
    for (int rt = 0; rt < 4; ++rt)
        for (int ct = 0; ct < 4; ++ct) acc[rt][ct] = vzero;

    auto stage = [&](int buf, int k0) {
        #pragma unroll
        for (int i = 0; i < 8; ++i) {            // A: 256x64 bf16 = 32 KB
            int flat = tid + i * 256;            // 0..2047
            int r = flat >> 3, c = (flat & 7) * 8;
            async_cp16(&As[buf][r * 64 + c], A + (size_t)(tileM + r) * lda + k0 + c);
        }
        #pragma unroll
        for (int i = 0; i < 4; ++i) {            // B: 128x64 bf16 = 16 KB
            int flat = tid + i * 256;            // 0..1023
            int r = flat >> 3, c = (flat & 7) * 8;
            int gn = tileN + r; if (gn >= N) gn = N - 1;   // clamp (stores guarded)
            async_cp16(&Bs[buf][r * 64 + c], BT + (size_t)gn * ldbt + k0 + c);
        }
    };

    stage(0, 0);
    int it = 0;
    for (int k0 = 0; k0 < K; k0 += 64, ++it) {
        const bool more = (k0 + 64 < K);
        if (more) stage((it + 1) % 3, k0 + 64);
        if (more) wait_async<12>(); else wait_async<0>();
        __syncthreads();
        const int buf = it % 3;
        const unsigned short* abase = As[buf] + (wm * 64) * 64;
        const unsigned short* bbase = Bs[buf] + (wn * 64) * 64;
        #pragma unroll
        for (int kk = 0; kk < 64; kk += 32) {
            v16bf af[4];
            #pragma unroll
            for (int rt = 0; rt < 4; ++rt)
                af[rt] = load_frag(abase + rt * 16 * 64, 64, lane, kk);
            #pragma unroll
            for (int ct = 0; ct < 4; ++ct) {
                v16bf bfg = load_frag(bbase + ct * 16 * 64, 64, lane, kk);
                #pragma unroll
                for (int rt = 0; rt < 4; ++rt)
                    acc[rt][ct] = wmma_bf16(af[rt], bfg, acc[rt][ct]);
            }
        }
    }

    const int half = lane >> 4;
    const int cj   = lane & 15;
    for (int rt = 0; rt < 4; ++rt)
        for (int ct = 0; ct < 4; ++ct) {
            int row = tileM + wm * 64 + rt * 16 + 8 * half;
            int col = tileN + wn * 64 + ct * 16 + cj;
            if (col < N) {
                float* cp = C + (size_t)row * ldc + coff + col;
                #pragma unroll
                for (int r = 0; r < 8; ++r) cp[(size_t)r * ldc] = acc[rt][ct][r];
            }
        }
}

// ---------------- causal depthwise conv (K=4) + SiLU -> bf16 ----------------
__global__ __launch_bounds__(256) void conv_silu_kernel(
    const float* __restrict__ xBC, const float* __restrict__ w,
    const float* __restrict__ bias, unsigned short* __restrict__ out)
{
    size_t id = (size_t)blockIdx.x * 256 + threadIdx.x;
    if (id >= (size_t)kB * kS * kCD) return;
    int ch    = (int)(id % kCD);
    size_t bs = id / kCD;
    int s     = (int)(bs % kS);
    float acc = bias[ch];
    #pragma unroll
    for (int k = 0; k < 4; ++k) {
        int ss = s - 3 + k;
        if (ss >= 0) acc += w[ch * 4 + k] * xBC[(bs - 3 + (size_t)k) * kCD + ch];
    }
    out[id] = f2bf(acc / (1.f + __expf(-acc)));
}

// ---------------- dt = softplus(raw + bias); cum = cumsum(dt * A) ----------------
__global__ __launch_bounds__(64) void dt_cumsum_kernel(
    const float* __restrict__ dtraw, const float* __restrict__ dt_bias,
    const float* __restrict__ A_log, float* __restrict__ dtw, float* __restrict__ cumw)
{
    int bc = blockIdx.x;
    int h  = threadIdx.x;
    int b  = bc / kNC, c = bc % kNC;
    float bias = dt_bias[h];
    float Ah   = -__expf(A_log[h]);
    float cum  = 0.f;
    int obase  = (bc * kH + h) * kL;
    size_t rbase = ((size_t)b * kS + c * kL) * kH + h;
    for (int l = 0; l < kL; ++l) {
        float r = dtraw[rbase + (size_t)l * kH] + bias;
        float v = (r > 20.f) ? r : log1pf(__expf(r));
        dtw[obase + l] = v;
        cum += v * Ah;
        cumw[obase + l] = cum;
    }
}

// ---------------- chunk kernel 1: intra-chunk attention + chunk states ----------------
__global__ __launch_bounds__(256) void chunk1_kernel(
    const unsigned short* __restrict__ xBCc, const float* __restrict__ dtw,
    const float* __restrict__ cumw, const float* __restrict__ Dp,
    float* __restrict__ y, float* __restrict__ states)
{
    __shared__ __align__(16) unsigned short sC[kL * kGN];    // C tile, later CB*Ldec
    __shared__ __align__(16) unsigned short sB[kL * kGN];    // B tile [l][n]
    __shared__ __align__(16) unsigned short sBTw[kGN * kL];  // (B^T * w) [n][l]
    __shared__ __align__(16) unsigned short sXT[kP * kL];    // x^T [p][l]
    __shared__ float scum[kL];
    __shared__ float sdt[kL];
    __shared__ float sw[kL];

    const int tid  = threadIdx.x;
    const int lane = tid & 31;
    const int wave = tid >> 5;
    const int bid  = blockIdx.x;
    const int h = bid % kH;
    const int c = (bid / kH) % kNC;
    const int b = bid / (kH * kNC);
    const size_t sbase = (size_t)b * kS + c * kL;
    const int chbase = ((b * kNC + c) * kH + h) * kL;

    if (tid < kL) { scum[tid] = cumw[chbase + tid]; sdt[tid] = dtw[chbase + tid]; }
    __syncthreads();
    if (tid < kL) sw[tid] = __expf(scum[kL - 1] - scum[tid]) * sdt[tid];

    // async-stage B and C tiles (rows are 256B contiguous in memory)
    #pragma unroll
    for (int i = 0; i < 8; ++i) {
        int flat = tid + i * 256;              // 0..2047
        int l = flat >> 4, cc = (flat & 15) * 8;
        async_cp16(&sB[l * kGN + cc], xBCc + (sbase + l) * kCD + kI + cc);
        async_cp16(&sC[l * kGN + cc], xBCc + (sbase + l) * kCD + kI + kGN + cc);
    }
    // x^T (needs layout transpose -> scalar ds stores)
    #pragma unroll 4
    for (int i = 0; i < 32; ++i) {
        int f = tid + i * 256;
        int p = f & (kP - 1), l = f >> 6;
        sXT[p * kL + l] = xBCc[(sbase + l) * kCD + h * kP + p];
    }
    wait_async<0>();
    __syncthreads();
    // B^T * w from LDS
    #pragma unroll 4
    for (int i = 0; i < 64; ++i) {
        int f = tid + i * 256;
        int n = f & (kGN - 1), l = f >> 7;
        sBTw[n * kL + l] = f2bf(bf2f(sB[l * kGN + n]) * sw[l]);
    }
    __syncthreads();

    // GEMM1: CB[i][j] = sum_n C[i][n] * B[j][n]
    const v8f vzero = {0.f,0.f,0.f,0.f,0.f,0.f,0.f,0.f};
    v8f acc1[8];
    #pragma unroll
    for (int ct = 0; ct < 8; ++ct) acc1[ct] = vzero;
    {
        const unsigned short* arow = sC + wave * 16 * kGN;
        #pragma unroll
        for (int kk = 0; kk < kGN; kk += 32) {
            v16bf af = load_frag(arow, kGN, lane, kk);
            #pragma unroll
            for (int ct = 0; ct < 8; ++ct) {
                v16bf bfg = load_frag(sB + ct * 16 * kGN, kGN, lane, kk);
                acc1[ct] = wmma_bf16(af, bfg, acc1[ct]);
            }
        }
    }
    __syncthreads();
    // decay mask + dt, overwrite sC with CB*Ldec (bf16)
    const int half = lane >> 4;
    const int nj   = lane & 15;
    #pragma unroll
    for (int ct = 0; ct < 8; ++ct)
        #pragma unroll
        for (int r = 0; r < 8; ++r) {
            int i = wave * 16 + r + 8 * half;
            int j = ct * 16 + nj;
            float d = 0.f;
            if (i >= j) d = acc1[ct][r] * __expf(scum[i] - scum[j]) * sdt[j];
            sC[i * kGN + j] = f2bf(d);
        }
    __syncthreads();

    // GEMM2: y_intra[i][p] = sum_j CBd[i][j] * xT[p][j]
    // GEMM3: statesT[n][p] = sum_l BTw[n][l] * xT[p][l]
    v8f acc2[4], acc3[4];
    #pragma unroll
    for (int ct = 0; ct < 4; ++ct) { acc2[ct] = vzero; acc3[ct] = vzero; }
    {
        const unsigned short* a2 = sC   + wave * 16 * kGN;
        const unsigned short* a3 = sBTw + wave * 16 * kL;
        #pragma unroll
        for (int kk = 0; kk < kL; kk += 32) {
            v16bf af2 = load_frag(a2, kGN, lane, kk);
            v16bf af3 = load_frag(a3, kL, lane, kk);
            #pragma unroll
            for (int ct = 0; ct < 4; ++ct) {
                v16bf bfg = load_frag(sXT + ct * 16 * kL, kL, lane, kk);
                acc2[ct] = wmma_bf16(af2, bfg, acc2[ct]);
                acc3[ct] = wmma_bf16(af3, bfg, acc3[ct]);
            }
        }
    }

    const float dpv = Dp[h];
    const size_t stbase = (size_t)((b * kNC + c) * kH + h) * (kP * kGN);
    #pragma unroll
    for (int ct = 0; ct < 4; ++ct)
        #pragma unroll
        for (int r = 0; r < 8; ++r) {
            int row = wave * 16 + r + 8 * half;   // l for acc2, n for acc3
            int p   = ct * 16 + nj;
            float xv = bf2f(sXT[p * kL + row]);
            y[(sbase + row) * kI + h * kP + p] = acc2[ct][r] + dpv * xv;
            states[stbase + (size_t)p * kGN + row] = acc3[ct][r];
        }
}

// ---------------- inter-chunk scan (sequential over chunks) ----------------
__global__ __launch_bounds__(256) void scan_kernel(
    const float* __restrict__ states, const float* __restrict__ cumw,
    unsigned short* __restrict__ prev)
{
    int bid = blockIdx.x;
    int h = bid % kH, b = bid / kH;
    int t = threadIdx.x;
    float carry[32];
    #pragma unroll
    for (int i = 0; i < 32; ++i) carry[i] = 0.f;
    for (int c = 0; c < kNC; ++c) {
        int chidx = (b * kNC + c) * kH + h;
        size_t base = (size_t)chidx * (kP * kGN);
        float dec = __expf(cumw[chidx * kL + (kL - 1)]);
        #pragma unroll
        for (int i = 0; i < 32; ++i) {
            int idx = t + i * 256;
            float st = states[base + idx];
            prev[base + idx] = f2bf(carry[i]);
            carry[i] = dec * carry[i] + st;
        }
    }
}

// ---------------- chunk kernel 2: y += exp(cum) * (C @ prev^T) ----------------
__global__ __launch_bounds__(256) void chunk2_kernel(
    const unsigned short* __restrict__ xBCc, const float* __restrict__ cumw,
    const unsigned short* __restrict__ prev, float* __restrict__ y)
{
    __shared__ __align__(16) unsigned short sC2[kL * kGN];   // C tile [l][n]
    __shared__ __align__(16) unsigned short sPrev[kP * kGN]; // prev [p][n]
    __shared__ float scum[kL];

    const int tid  = threadIdx.x;
    const int lane = tid & 31;
    const int wave = tid >> 5;
    const int bid  = blockIdx.x;
    const int h = bid % kH;
    const int c = (bid / kH) % kNC;
    const int b = bid / (kH * kNC);
    const size_t sbase = (size_t)b * kS + c * kL;
    const int chbase = ((b * kNC + c) * kH + h) * kL;
    const size_t pbase = (size_t)((b * kNC + c) * kH + h) * (kP * kGN);

    if (tid < kL) scum[tid] = cumw[chbase + tid];
    #pragma unroll
    for (int i = 0; i < 8; ++i) {
        int flat = tid + i * 256;
        int l = flat >> 4, cc = (flat & 15) * 8;
        async_cp16(&sC2[l * kGN + cc], xBCc + (sbase + l) * kCD + kI + kGN + cc);
    }
    #pragma unroll
    for (int i = 0; i < 4; ++i) {
        int flat = tid + i * 256;               // 0..1023 chunks of 8 elems
        async_cp16(&sPrev[flat * 8], prev + pbase + flat * 8);
    }
    wait_async<0>();
    __syncthreads();

    const v8f vzero = {0.f,0.f,0.f,0.f,0.f,0.f,0.f,0.f};
    v8f acc[4];
    #pragma unroll
    for (int ct = 0; ct < 4; ++ct) acc[ct] = vzero;
    const unsigned short* arow = sC2 + wave * 16 * kGN;
    #pragma unroll
    for (int kk = 0; kk < kGN; kk += 32) {
        v16bf af = load_frag(arow, kGN, lane, kk);
        #pragma unroll
        for (int ct = 0; ct < 4; ++ct) {
            v16bf bfg = load_frag(sPrev + ct * 16 * kGN, kGN, lane, kk);
            acc[ct] = wmma_bf16(af, bfg, acc[ct]);
        }
    }

    const int half = lane >> 4;
    const int nj   = lane & 15;
    #pragma unroll
    for (int ct = 0; ct < 4; ++ct)
        #pragma unroll
        for (int r = 0; r < 8; ++r) {
            int l = wave * 16 + r + 8 * half;
            int p = ct * 16 + nj;
            size_t idx = (sbase + l) * kI + h * kP + p;
            y[idx] += __expf(scum[l]) * acc[ct][r];
        }
}

// ---------------- gated RMSNorm -> bf16 ----------------
__global__ __launch_bounds__(256) void rmsnorm_kernel(
    float* __restrict__ y, const float* __restrict__ gate,
    const float* __restrict__ nw, unsigned short* __restrict__ ybf)
{
    __shared__ float red[256];
    size_t row = blockIdx.x;
    int t = threadIdx.x;
    float ss = 0.f;
    #pragma unroll 4
    for (int j = 0; j < 16; ++j) {
        int i = t + j * 256;
        size_t idx = row * kI + i;
        float v = y[idx];
        float g = gate[idx];
        v = v * (g / (1.f + __expf(-g)));
        y[idx] = v;
        ss += v * v;
    }
    red[t] = ss;
    __syncthreads();
    for (int s = 128; s > 0; s >>= 1) {
        if (t < s) red[t] += red[t + s];
        __syncthreads();
    }
    float scale = rsqrtf(red[0] / (float)kI + kEPS);
    #pragma unroll 4
    for (int j = 0; j < 16; ++j) {
        int i = t + j * 256;
        size_t idx = row * kI + i;
        ybf[idx] = f2bf(y[idx] * scale * nw[i]);
    }
}

// ---------------- host launcher ----------------
extern "C" void kernel_launch(void* const* d_in, const int* in_sizes, int n_in,
                              void* d_out, int out_size, void* d_ws, size_t ws_size,
                              hipStream_t stream) {
    (void)in_sizes; (void)n_in; (void)out_size; (void)ws_size;
    const float* hid     = (const float*)d_in[0];
    const float* Wz      = (const float*)d_in[1];
    const float* Wx      = (const float*)d_in[2];
    const float* WB      = (const float*)d_in[3];
    const float* WC      = (const float*)d_in[4];
    const float* Wdt     = (const float*)d_in[5];
    const float* conv_w  = (const float*)d_in[6];
    const float* conv_b  = (const float*)d_in[7];
    const float* dt_bias = (const float*)d_in[8];
    const float* A_log   = (const float*)d_in[9];
    const float* Dp      = (const float*)d_in[10];
    const float* norm_w  = (const float*)d_in[11];
    const float* Wout    = (const float*)d_in[12];
    float* out = (float*)d_out;

    const size_t BS = (size_t)kB * kS;              // 4096
    // fp32 scratch
    float* fp = (float*)d_ws;
    float* gate  = fp; fp += BS * kI;
    float* xbc   = fp; fp += BS * kCD;
    float* dtraw = fp; fp += BS * kH;
    float* dtw   = fp; fp += (size_t)kB * kNC * kH * kL;
    float* cumw  = fp; fp += (size_t)kB * kNC * kH * kL;
    float* yb    = fp; fp += BS * kI;
    float* st    = fp; fp += (size_t)kB * kNC * kH * kP * kGN;
    // bf16 scratch
    unsigned short* up = (unsigned short*)fp;
    unsigned short* hid_bf = up; up += BS * kD;
    unsigned short* WzT  = up; up += (size_t)kI * kD;
    unsigned short* WxT  = up; up += (size_t)kI * kD;
    unsigned short* WBT  = up; up += (size_t)kGN * kD;
    unsigned short* WCT  = up; up += (size_t)kGN * kD;
    unsigned short* WdtT = up; up += (size_t)kH * kD;
    unsigned short* WoT  = up; up += (size_t)kD * kI;
    unsigned short* xbcc = up; up += BS * kCD;
    unsigned short* pv   = up; up += (size_t)kB * kNC * kH * kP * kGN;
    unsigned short* ybf  = up; up += BS * kI;

    dim3 blk(256);
    // pre-pass: casts + weight transposes
    cast_bf16_kernel<<<dim3((unsigned)((BS * kD / 4 + 255) / 256)), blk, 0, stream>>>(
        hid, hid_bf, BS * kD / 4);
    transpose_bf16_kernel<<<dim3(kI / 32, kD / 32), blk, 0, stream>>>(Wz, WzT, kD, kI);
    transpose_bf16_kernel<<<dim3(kI / 32, kD / 32), blk, 0, stream>>>(Wx, WxT, kD, kI);
    transpose_bf16_kernel<<<dim3(kGN / 32, kD / 32), blk, 0, stream>>>(WB, WBT, kD, kGN);
    transpose_bf16_kernel<<<dim3(kGN / 32, kD / 32), blk, 0, stream>>>(WC, WCT, kD, kGN);
    transpose_bf16_kernel<<<dim3(kH / 32, kD / 32), blk, 0, stream>>>(Wdt, WdtT, kD, kH);
    transpose_bf16_kernel<<<dim3(kD / 32, kI / 32), blk, 0, stream>>>(Wout, WoT, kI, kD);

    // input projections (M = 4096 -> 16 row-blocks of 256)
    gemm_bf16_kernel<<<dim3(kI / 128, BS / 256), blk, 0, stream>>>(
        hid_bf, WzT, gate, (int)BS, kI, kD, kD, kD, kI, 0);
    gemm_bf16_kernel<<<dim3(kI / 128, BS / 256), blk, 0, stream>>>(
        hid_bf, WxT, xbc, (int)BS, kI, kD, kD, kD, kCD, 0);
    gemm_bf16_kernel<<<dim3(1, BS / 256), blk, 0, stream>>>(
        hid_bf, WBT, xbc, (int)BS, kGN, kD, kD, kD, kCD, kI);
    gemm_bf16_kernel<<<dim3(1, BS / 256), blk, 0, stream>>>(
        hid_bf, WCT, xbc, (int)BS, kGN, kD, kD, kD, kCD, kI + kGN);
    gemm_bf16_kernel<<<dim3(1, BS / 256), blk, 0, stream>>>(
        hid_bf, WdtT, dtraw, (int)BS, kH, kD, kD, kD, kH, 0);

    // causal conv + silu -> bf16
    size_t convN = BS * kCD;
    conv_silu_kernel<<<dim3((unsigned)((convN + 255) / 256)), blk, 0, stream>>>(
        xbc, conv_w, conv_b, xbcc);

    // dt softplus + per-chunk cumsum
    dt_cumsum_kernel<<<dim3(kB * kNC), dim3(kH), 0, stream>>>(
        dtraw, dt_bias, A_log, dtw, cumw);

    // intra-chunk attention + chunk end-states
    chunk1_kernel<<<dim3(kB * kNC * kH), blk, 0, stream>>>(
        xbcc, dtw, cumw, Dp, yb, st);

    // inter-chunk recurrence (prev stored bf16)
    scan_kernel<<<dim3(kB * kH), blk, 0, stream>>>(st, cumw, pv);

    // inter-chunk contribution
    chunk2_kernel<<<dim3(kB * kNC * kH), blk, 0, stream>>>(xbcc, cumw, pv, yb);

    // gated RMSNorm -> bf16
    rmsnorm_kernel<<<dim3((unsigned)BS), blk, 0, stream>>>(yb, gate, norm_w, ybf);

    // output projection
    gemm_bf16_kernel<<<dim3(kD / 128, BS / 256), blk, 0, stream>>>(
        ybf, WoT, out, (int)BS, kD, kI, kI, kI, kD, 0);
}
